// MultiHeadAttentionBlock_2920577762079
// MI455X (gfx1250) — matrix operand order repaired and checked by reading, not compiled, not run on hardware
//
#include <hip/hip_runtime.h>
#include <hip/hip_bf16.h>

typedef _Float16 v8h  __attribute__((ext_vector_type(8)));
typedef _Float16 v16h __attribute__((ext_vector_type(16)));
typedef float    v8f  __attribute__((ext_vector_type(8)));
typedef unsigned int u32x4 __attribute__((ext_vector_type(4)));
typedef int          i32x4 __attribute__((ext_vector_type(4)));
typedef int          i32x8 __attribute__((ext_vector_type(8)));

#define D_MODEL 1024
#define SEQ     2048
#define BATCH   2
#define NHEAD   16
#define DKH     64
#define MTOT    (BATCH * SEQ)   // 4096

// ---------------------------------------------------------------------------
// WMMA fragment loaders (contiguous K order shared by A and B, not the ISA table order; wave32)
// A 16x32 f16: lane L holds row (L&15); halves = K (L>>4)*16 .. +15
// contiguous (same K order as B).
// B 32x16 f16 (from a K-major buffer, row = output column n):
// lane L holds col (L&15); halves = K (L>>4)*16 .. +15 contiguous.
// ---------------------------------------------------------------------------
__device__ __forceinline__ v16h load_a_frag(const _Float16* __restrict__ base,
                                            int ldk, int lane) {
  const _Float16* p = base + (size_t)(lane & 15) * ldk + ((lane >> 4) << 4);
  v8h lo = *(const v8h*)p;
  v8h hi = *(const v8h*)(p + 8);
  v16h a;
#pragma unroll
  for (int i = 0; i < 8; ++i) { a[i] = lo[i]; a[8 + i] = hi[i]; }
  return a;
}

__device__ __forceinline__ v16h load_b_frag(const _Float16* __restrict__ base,
                                            int ldk, int lane) {
  const _Float16* p = base + (size_t)(lane & 15) * ldk + ((lane >> 4) << 4);
  return *(const v16h*)p;
}

// ---------------------------------------------------------------------------
// Tensor Data Mover: 2D f16 tile, global -> LDS (D# layout per ISA ch.8).
// One TDM op moves tile_d1 rows x tile_d0 elements (row stride in elements).
// ---------------------------------------------------------------------------
__device__ __forceinline__ void tdm_load_2d_f16(const _Float16* gsrc,
                                                uint32_t lds_byte_addr,
                                                uint32_t tile_d0,
                                                uint32_t tile_d1,
                                                uint32_t tensor_d0,
                                                uint32_t tensor_d1,
                                                uint64_t row_stride_elems) {
  uint64_t ga = (uint64_t)(size_t)gsrc;
  u32x4 g0;
  g0[0] = 1u;                                   // count=1, user descriptor
  g0[1] = lds_byte_addr;                        // lds_addr
  g0[2] = (uint32_t)ga;                         // global_addr[31:0]
  g0[3] = (uint32_t)((ga >> 32) & 0x01FFFFFFu)  // global_addr[56:32]
          | (2u << 30);                         // type = 2 ("image")
  i32x8 g1;
  g1[0] = (int)(1u << 16);                      // wg_mask=0, data_size=1 (2B)
  g1[1] = (int)((tensor_d0 & 0xFFFFu) << 16);   // tensor_dim0[15:0]
  g1[2] = (int)((tensor_d0 >> 16) | ((tensor_d1 & 0xFFFFu) << 16));
  g1[3] = (int)((tensor_d1 >> 16) | (tile_d0 << 16));   // tile_dim0
  g1[4] = (int)(tile_d1 & 0xFFFFu);             // tile_dim1 (tile_dim2 = 0)
  g1[5] = (int)(uint32_t)(row_stride_elems & 0xFFFFFFFFu);  // dim0_stride lo
  g1[6] = (int)(uint32_t)((row_stride_elems >> 32) & 0xFFFFu);
  g1[7] = 0;                                    // dim1_stride unused (2D tile)
  i32x4 z4 = {0, 0, 0, 0};
#if __clang_major__ >= 23
  i32x8 z8 = {0, 0, 0, 0, 0, 0, 0, 0};
  __builtin_amdgcn_tensor_load_to_lds(g0, g1, z4, z4, z8, 0);
#else
  __builtin_amdgcn_tensor_load_to_lds(g0, g1, z4, z4, 0);
#endif
}

// ---------------------------------------------------------------------------
// fp32 -> f16 elementwise cast
// ---------------------------------------------------------------------------
__global__ void cast_f16_kernel(const float* __restrict__ in,
                                _Float16* __restrict__ out, int n) {
  int i = blockIdx.x * blockDim.x + threadIdx.x;
  if (i < n) out[i] = (_Float16)in[i];
}

// W [K][N] fp32 -> Wt [N][K] f16  (K-major for contiguous B fragments)
__global__ void transpose_cast_kernel(const float* __restrict__ W,
                                      _Float16* __restrict__ Wt, int K, int N) {
  int n = blockIdx.x * 16 + threadIdx.x;
  int k = blockIdx.y * 16 + threadIdx.y;
  if (n < N && k < K) Wt[(size_t)n * K + k] = (_Float16)W[(size_t)k * N + n];
}

// ---------------------------------------------------------------------------
// WMMA GEMM: out[M][N] = A[M][K] * Bt[N][K]^T * oscale + bias (+ resid)
// One wave computes a 16x64 tile (A-fragment reused across 4 N-tiles).
// Block = 4 waves = 16 x 256 of output.
// ---------------------------------------------------------------------------
__global__ void __launch_bounds__(128)
wmma_gemm_kernel(const _Float16* __restrict__ A,
                 const _Float16* __restrict__ Bt,
                 const float* __restrict__ bias,
                 _Float16* __restrict__ outh,     // f16 output (or null)
                 float* __restrict__ outf,        // f32 output (or null)
                 const float* __restrict__ resid, // optional residual [M][N]
                 int Mdim, int Ndim, int Kd, float oscale) {
  const int lane = threadIdx.x & 31;
  const int wv   = threadIdx.x >> 5;
  const int m0   = blockIdx.x * 16;
  const int n0   = blockIdx.y * 256 + wv * 64;
  if (m0 >= Mdim || n0 >= Ndim) return;

  v8f acc[4];
#pragma unroll
  for (int t = 0; t < 4; ++t)
#pragma unroll
    for (int i = 0; i < 8; ++i) acc[t][i] = 0.0f;

  for (int k = 0; k < Kd; k += 32) {
    v16h a = load_a_frag(A + (size_t)m0 * Kd + k, Kd, lane);
#pragma unroll
    for (int t = 0; t < 4; ++t) {
      v16h b = load_b_frag(Bt + (size_t)(n0 + t * 16) * Kd + k, Kd, lane);
      acc[t] = __builtin_amdgcn_wmma_f32_16x16x32_f16(
          false, a, false, b, (short)0, acc[t], false, false);
    }
  }

  // C layout: VGPR r -> row r (lanes 0-15) / row r+8 (lanes 16-31), col = lane&15
  const int colL = lane & 15;
  const int rhi  = (lane >> 4) << 3;
#pragma unroll
  for (int t = 0; t < 4; ++t) {
    int n    = n0 + t * 16 + colL;
    float bv = bias ? bias[n] : 0.0f;
#pragma unroll
    for (int r = 0; r < 8; ++r) {
      int m      = m0 + r + rhi;
      size_t idx = (size_t)m * Ndim + n;
      float v    = acc[t][r] * oscale + bv;
      if (resid) v += resid[idx];
      if (outh) outh[idx] = (_Float16)v;
      else      outf[idx] = v;
    }
  }
}

// ---------------------------------------------------------------------------
// Flash attention: block = 4 waves sharing (b, h); each wave owns a 16-query
// tile. K chunk staged by one TDM tensor_load_to_lds per 32-key step; V chunk
// transposed into LDS cooperatively. Online softmax; Q pre-scaled by 1/8.
// ---------------------------------------------------------------------------
__global__ void __launch_bounds__(128)
flash_attn_kernel(const _Float16* __restrict__ Qh,
                  const _Float16* __restrict__ Kh,
                  const _Float16* __restrict__ Vh,
                  _Float16* __restrict__ Ch) {
  __shared__ __align__(32) _Float16 Ktile[32 * DKH];    // [key][d] (TDM dest)
  __shared__ __align__(32) _Float16 VtileT[DKH * 32];   // [d][key]
  __shared__ __align__(32) _Float16 Pbuf[4][16 * 32];   // per-wave P tile

  const int lane = threadIdx.x & 31;
  const int wv   = threadIdx.x >> 5;
  const int b    = blockIdx.z;
  const int h    = blockIdx.y;
  const int q0   = (blockIdx.x * 4 + wv) * 16;

  const size_t basebh = (size_t)b * SEQ * D_MODEL + (size_t)h * DKH;
  const uint32_t ktile_lds = (uint32_t)(size_t)&Ktile[0];

  // Q A-fragments: 16 x 64 in two K=32 slabs, reused for all key chunks
  const _Float16* qbase = Qh + basebh + (size_t)q0 * D_MODEL;
  v16h qa0 = load_a_frag(qbase + 0,  D_MODEL, lane);
  v16h qa1 = load_a_frag(qbase + 32, D_MODEL, lane);

  v8f ctx[4];
#pragma unroll
  for (int t = 0; t < 4; ++t)
#pragma unroll
    for (int i = 0; i < 8; ++i) ctx[t][i] = 0.0f;
  float mrun[8], lrun[8];
#pragma unroll
  for (int r = 0; r < 8; ++r) { mrun[r] = -3.0e38f; lrun[r] = 0.0f; }

  for (int kb = 0; kb < SEQ; kb += 32) {
    __syncthreads();  // previous chunk's readers done

    // K chunk: one TDM descriptor moves the whole 32x64 f16 tile to LDS
    if (wv == 0) {
      tdm_load_2d_f16(Kh + basebh + (size_t)kb * D_MODEL, ktile_lds,
                      /*tile_d0=*/DKH, /*tile_d1=*/32,
                      /*tensor_d0=*/DKH, /*tensor_d1=*/32,
                      /*row_stride=*/D_MODEL);
    }

    // V chunk: cooperative transpose stage [key][d] -> [d][key]
    for (int idx = threadIdx.x; idx < 32 * 32; idx += 128) {
      int row = idx >> 5;   // key within chunk
      int dw  = idx & 31;   // dword index along d (2 halves)
      const _Float16* vp = Vh + basebh + (size_t)(kb + row) * D_MODEL + dw * 2;
      VtileT[(dw * 2 + 0) * 32 + row] = vp[0];
      VtileT[(dw * 2 + 1) * 32 + row] = vp[1];
    }

    if (wv == 0) __builtin_amdgcn_s_wait_tensorcnt(0);
    __syncthreads();  // K (TDM) + V (ds stores) visible to all waves

    if (kb + 32 < SEQ)  // prefetch next K chunk (one row per lane)
      __builtin_prefetch(Kh + basebh + (size_t)(kb + 32 + lane) * D_MODEL, 0, 1);

    // S = Q * K^T : two 16x16 score tiles (keys kb..+15, kb+16..+31)
    v8f sc[2];
#pragma unroll
    for (int t = 0; t < 2; ++t) {
#pragma unroll
      for (int i = 0; i < 8; ++i) sc[t][i] = 0.0f;
#pragma unroll
      for (int j = 0; j < 2; ++j) {  // d slabs 0..31 / 32..63
        v16h bk = *(const v16h*)(Ktile + (size_t)(t * 16 + (lane & 15)) * DKH +
                                 j * 32 + ((lane >> 4) << 4));
        sc[t] = __builtin_amdgcn_wmma_f32_16x16x32_f16(
            false, (j ? qa1 : qa0), false, bk, (short)0, sc[t], false, false);
      }
    }

    // Online softmax per row (row r of tile lives in lanes of one 16-half)
    const int colL = lane & 15;
    const int rhi  = (lane >> 4) << 3;
#pragma unroll
    for (int r = 0; r < 8; ++r) {
      float mx = fmaxf(sc[0][r], sc[1][r]);
#pragma unroll
      for (int mask = 1; mask <= 8; mask <<= 1)
        mx = fmaxf(mx, __shfl_xor(mx, mask, 32));
      float mnew  = fmaxf(mrun[r], mx);
      float alpha = __expf(mrun[r] - mnew);
      float p0    = __expf(sc[0][r] - mnew);
      float p1    = __expf(sc[1][r] - mnew);
      float rs    = p0 + p1;
#pragma unroll
      for (int mask = 1; mask <= 8; mask <<= 1)
        rs += __shfl_xor(rs, mask, 32);
      lrun[r] = lrun[r] * alpha + rs;
      mrun[r] = mnew;
      int row = r + rhi;
      Pbuf[wv][row * 32 + colL]      = (_Float16)p0;
      Pbuf[wv][row * 32 + 16 + colL] = (_Float16)p1;
#pragma unroll
      for (int t = 0; t < 4; ++t) ctx[t][r] *= alpha;
    }

    // LDS RAW within this wave: drain DScnt before re-reading P
    asm volatile("s_wait_dscnt 0" ::: "memory");

    // P (C layout) re-read as A 16x32 fragment
    v16h pa = load_a_frag(&Pbuf[wv][0], 32, lane);

    // ctx += P * V  (4 dim-tiles of 16)
#pragma unroll
    for (int t = 0; t < 4; ++t) {
      v16h bv = *(const v16h*)(VtileT + (size_t)(t * 16 + (lane & 15)) * 32 +
                               ((lane >> 4) << 4));
      ctx[t] = __builtin_amdgcn_wmma_f32_16x16x32_f16(
          false, pa, false, bv, (short)0, ctx[t], false, false);
    }
  }

  // Normalize and store ctx -> [b, s, h*dk] f16
  const int colL = lane & 15;
  const int rhi  = (lane >> 4) << 3;
#pragma unroll
  for (int t = 0; t < 4; ++t)
#pragma unroll
    for (int r = 0; r < 8; ++r) {
      int qrow = q0 + r + rhi;
      int d    = t * 16 + colL;
      Ch[(size_t)(b * SEQ + qrow) * D_MODEL + (size_t)h * DKH + d] =
          (_Float16)(ctx[t][r] / lrun[r]);
    }
}

// ---------------------------------------------------------------------------
// Host-side orchestration
// ---------------------------------------------------------------------------
extern "C" void kernel_launch(void* const* d_in, const int* in_sizes, int n_in,
                              void* d_out, int out_size, void* d_ws,
                              size_t ws_size, hipStream_t stream) {
  const float* x  = (const float*)d_in[0];
  const float* Wq = (const float*)d_in[1];
  const float* bq = (const float*)d_in[2];
  const float* Wk = (const float*)d_in[3];
  const float* bk = (const float*)d_in[4];
  const float* Wv = (const float*)d_in[5];
  const float* bv = (const float*)d_in[6];
  const float* Wo = (const float*)d_in[7];
  const float* bo = (const float*)d_in[8];
  float* out = (float*)d_out;

  // Workspace layout (f16 halves): ~48 MiB total
  _Float16* ws  = (_Float16*)d_ws;
  const size_t MD = (size_t)MTOT * D_MODEL;     // 4M
  const size_t DD = (size_t)D_MODEL * D_MODEL;  // 1M
  _Float16* xh  = ws;
  _Float16* wqt = xh + MD;
  _Float16* wkt = wqt + DD;
  _Float16* wvt = wkt + DD;
  _Float16* wot = wvt + DD;
  _Float16* qh  = wot + DD;
  _Float16* kh  = qh + MD;
  _Float16* vh  = kh + MD;
  _Float16* ch  = vh + MD;

  // 1) casts
  {
    int n = (int)MD;
    cast_f16_kernel<<<(n + 255) / 256, 256, 0, stream>>>(x, xh, n);
    dim3 g(D_MODEL / 16, D_MODEL / 16), blk(16, 16);
    transpose_cast_kernel<<<g, blk, 0, stream>>>(Wq, wqt, D_MODEL, D_MODEL);
    transpose_cast_kernel<<<g, blk, 0, stream>>>(Wk, wkt, D_MODEL, D_MODEL);
    transpose_cast_kernel<<<g, blk, 0, stream>>>(Wv, wvt, D_MODEL, D_MODEL);
    transpose_cast_kernel<<<g, blk, 0, stream>>>(Wo, wot, D_MODEL, D_MODEL);
  }

  // 2) Q/K/V projections (Q pre-scaled by 1/sqrt(dk) = 0.125)
  {
    dim3 g(MTOT / 16, D_MODEL / 256);
    wmma_gemm_kernel<<<g, 128, 0, stream>>>(xh, wqt, bq, qh, nullptr, nullptr,
                                            MTOT, D_MODEL, D_MODEL, 0.125f);
    wmma_gemm_kernel<<<g, 128, 0, stream>>>(xh, wkt, bk, kh, nullptr, nullptr,
                                            MTOT, D_MODEL, D_MODEL, 1.0f);
    wmma_gemm_kernel<<<g, 128, 0, stream>>>(xh, wvt, bv, vh, nullptr, nullptr,
                                            MTOT, D_MODEL, D_MODEL, 1.0f);
  }

  // 3) attention
  {
    dim3 g(SEQ / 64, NHEAD, BATCH);  // 4 waves x 16 queries per block
    flash_attn_kernel<<<g, 128, 0, stream>>>(qh, kh, vh, ch);
  }

  // 4) output projection + bias + residual (fp32 out)
  {
    dim3 g(MTOT / 16, D_MODEL / 256);
    wmma_gemm_kernel<<<g, 128, 0, stream>>>(ch, wot, bo, nullptr, out, x,
                                            MTOT, D_MODEL, D_MODEL, 1.0f);
  }
}
